// LSTMscratch_76055280877768
// MI455X (gfx1250) — matrix operand order repaired and checked by reading, not compiled, not run on hardware
//
#include <hip/hip_runtime.h>
#include <hip/hip_bf16.h>

// ---------------------------------------------------------------------------
// CDNA5 (gfx1250) implementation: fused conv pipeline + WMMA GEMMs + LSTM.
//
// Sizes (from reference): B=32, T=32, IMG=64, D_IN=256, HID=512, N_CLS=6
//   frames F = B*T = 1024
//   z1: (F,16,32,32)  z2: (F,32,16,16)   [kept in LDS only]
//   S : (F, 288) f16   emb: (F, 256) f16
//   W0t: (2048, 768) f16   W1t: (2048, 1024) f16   W3h: (256, 288) f16
// ---------------------------------------------------------------------------

typedef __attribute__((ext_vector_type(16))) _Float16 v16h;
typedef __attribute__((ext_vector_type(8)))  float    v8f;

union AFrag { v16h v; unsigned int u[8]; };

// A-matrix 16x32 f16 fragment (contiguous k = 16*half + s, same order as B): lane half = lane>>4, m = lane&15.
// dword j in 0..3 : K = half*16 + 2j       (VGPR 0..3)
// dword j in 4..7 : K = half*16 + 8 + 2(j-4)  (VGPR 4..7)
__device__ __forceinline__ v16h load_a_frag(const _Float16* __restrict__ rowPtr, int half) {
  AFrag a;
#pragma unroll
  for (int j = 0; j < 4; ++j)
    a.u[j] = *(const unsigned int*)(rowPtr + half * 16 + 2 * j);
#pragma unroll
  for (int j = 0; j < 4; ++j)
    a.u[4 + j] = *(const unsigned int*)(rowPtr + half * 16 + 8 + 2 * j);
  return a.v;
}

// B-matrix 32x16 f16 fragment: lane n = lane&15, K base = (lane>>4)*16,
// 16 consecutive K values -> contiguous 32 bytes in N-major (N x K) storage.
__device__ __forceinline__ v16h load_b_frag(const _Float16* __restrict__ p) {
  return *(const v16h*)p;  // 32B aligned by construction -> 2x global_load_b128
}

// ---------------------------------------------------------------------------
// Kernel 1: per-frame fused conv1+relu+pool, conv2+relu+pool, windowed sums.
// One block per frame; x frame, z1, z2 all staged in LDS (112 KB dynamic).
// Writes Sbuf[f*288 + c*9 + ky*3 + kx] = S[c,ky,kx] / 256  (f16)
// ---------------------------------------------------------------------------
__global__ void frames_kernel(const float* __restrict__ x,
                              const float* __restrict__ w1, const float* __restrict__ b1,
                              const float* __restrict__ w2, const float* __restrict__ b2,
                              _Float16* __restrict__ Sbuf) {
  extern __shared__ float smem[];
  float* xin = smem;                 // 64*64      = 4096 floats
  float* z1  = smem + 4096;          // 16*32*32   = 16384 floats
  float* z2  = smem + 4096 + 16384;  // 32*16*16   = 8192 floats

  const int f   = blockIdx.x;
  const int tid = threadIdx.x;

  const float* xf = x + (size_t)f * 4096;
  for (int i = tid; i < 4096; i += 256) xin[i] = xf[i];
  __syncthreads();

  // ---- conv1 (1->16, 3x3 SAME) + relu + 2x2 maxpool : 64x64 -> 32x32 ----
  for (int o = tid; o < 16384; o += 256) {
    const int c  = o >> 10;
    const int rem = o & 1023;
    const int py = rem >> 5, px = rem & 31;
    const float* wc = w1 + c * 9;
    const float  bc = b1[c];
    float best = 0.f;  // relu outputs are >= 0
#pragma unroll
    for (int dy = 0; dy < 2; ++dy)
#pragma unroll
      for (int dx = 0; dx < 2; ++dx) {
        const int y = 2 * py + dy, xx = 2 * px + dx;
        float acc = bc;
#pragma unroll
        for (int ky = 0; ky < 3; ++ky) {
          const int iy = y + ky - 1;
          if ((unsigned)iy < 64u) {
#pragma unroll
            for (int kx = 0; kx < 3; ++kx) {
              const int ix = xx + kx - 1;
              if ((unsigned)ix < 64u) acc = fmaf(wc[ky * 3 + kx], xin[iy * 64 + ix], acc);
            }
          }
        }
        best = fmaxf(best, acc);
      }
    z1[o] = best;
  }
  __syncthreads();

  // ---- conv2 (16->32, 3x3 SAME) + relu + 2x2 maxpool : 32x32 -> 16x16 ----
  for (int o = tid; o < 8192; o += 256) {
    const int c2  = o >> 8;
    const int rem = o & 255;
    const int qy = rem >> 4, qx = rem & 15;
    const float bc = b2[c2];
    float best = 0.f;
#pragma unroll
    for (int dy = 0; dy < 2; ++dy)
#pragma unroll
      for (int dx = 0; dx < 2; ++dx) {
        const int y = 2 * qy + dy, xx = 2 * qx + dx;
        float acc = bc;
        for (int ci = 0; ci < 16; ++ci) {
          const float* wp = w2 + (c2 * 16 + ci) * 9;
          const float* zp = z1 + ci * 1024;
#pragma unroll
          for (int ky = 0; ky < 3; ++ky) {
            const int iy = y + ky - 1;
            if ((unsigned)iy < 32u) {
#pragma unroll
              for (int kx = 0; kx < 3; ++kx) {
                const int ix = xx + kx - 1;
                if ((unsigned)ix < 32u) acc = fmaf(wp[ky * 3 + kx], zp[iy * 32 + ix], acc);
              }
            }
          }
        }
        best = fmaxf(best, acc);
      }
    z2[o] = best;
  }
  __syncthreads();

  // ---- 9 windowed sums per channel (= spatial mean of SAME conv3 taps) ----
  if (tid < 32) {
    const float* zc = z2 + tid * 256;
    float tot = 0.f, r0 = 0.f, r15 = 0.f, cl = 0.f, cr = 0.f;
    for (int i = 0; i < 256; ++i) {
      const int yy = i >> 4, xx = i & 15;
      const float v = zc[i];
      tot += v;
      if (yy == 0)  r0  += v;
      if (yy == 15) r15 += v;
      if (xx == 0)  cl  += v;
      if (xx == 15) cr  += v;
    }
    const float z00 = zc[0], z0r = zc[15], zr0 = zc[240], zrr = zc[255];
    // tap ky: 0 excludes row15, 1 full, 2 excludes row0 (same for kx/cols)
    const float R[3] = {r15, 0.f, r0};
    const float C[3] = {cr, 0.f, cl};
    _Float16* sp = Sbuf + (size_t)f * 288 + tid * 9;
#pragma unroll
    for (int ky = 0; ky < 3; ++ky)
#pragma unroll
      for (int kx = 0; kx < 3; ++kx) {
        float corner = 0.f;
        if (ky == 0 && kx == 0) corner = zrr;
        else if (ky == 0 && kx == 2) corner = zr0;
        else if (ky == 2 && kx == 0) corner = z0r;
        else if (ky == 2 && kx == 2) corner = z00;
        const float s = tot - R[ky] - C[kx] + corner;
        sp[ky * 3 + kx] = (_Float16)(s * (1.0f / 256.0f));
      }
  }
}

// ---------------------------------------------------------------------------
// Kernel 2: emb GEMM via WMMA: (1024 x 288) f16 @ (288 x 256) + b3 -> f16
// Wave computes a 16x64 tile (4 accumulators). 256 waves total.
// ---------------------------------------------------------------------------
__global__ void emb_gemm_kernel(const _Float16* __restrict__ S,
                                const _Float16* __restrict__ W3h,   // (256 x 288) N-major
                                const float* __restrict__ b3,
                                _Float16* __restrict__ emb) {
  const int tid  = threadIdx.x;
  const int lane = tid & 31;
  const int wid  = tid >> 5;
  const int gw   = blockIdx.x * 8 + wid;   // 0..255
  const int m0   = (gw >> 2) * 16;
  const int n0   = (gw & 3) * 64;
  const int half = lane >> 4;
  const int nlo  = lane & 15;
  const int row  = m0 + nlo;

  v8f acc[4] = {};
  for (int k0 = 0; k0 < 288; k0 += 32) {
    const v16h a = load_a_frag(S + (size_t)row * 288 + k0, half);
#pragma unroll
    for (int nt = 0; nt < 4; ++nt) {
      const v16h b = load_b_frag(W3h + (size_t)(n0 + nt * 16 + nlo) * 288 + k0 + half * 16);
      acc[nt] = __builtin_amdgcn_wmma_f32_16x16x32_f16(false, a, false, b, (short)0,
                                                       acc[nt], false, false);
    }
  }
#pragma unroll
  for (int nt = 0; nt < 4; ++nt) {
    const int n  = n0 + nt * 16 + nlo;
    const float bb = b3[n];
#pragma unroll
    for (int r = 0; r < 8; ++r) {
      const int m = m0 + r + 8 * half;   // D layout: M = r + 8*(lane>=16)
      emb[(size_t)m * 256 + n] = (_Float16)(acc[nt][r] + bb);
    }
  }
}

// ---------------------------------------------------------------------------
// Kernel 3: LSTM gate GEMM via WMMA.
//   gates(32 x 2048) = [Ax | Ah] (32 x Ktot) @ Wt^T + bias
//   A row m: k <  KX -> aX + m*strideX + k
//            k >= KX -> aH + m*KH + (k - KX)
// 64 waves (2 M-tiles x 32 N-groups of 64).
// ---------------------------------------------------------------------------
__global__ void gate_gemm_kernel(const _Float16* __restrict__ aX, int strideX, int KX,
                                 const _Float16* __restrict__ aH, int KH,
                                 const _Float16* __restrict__ Wt, int Ktot,  // (2048 x Ktot)
                                 const float* __restrict__ bias,
                                 float* __restrict__ gates) {
  const int tid  = threadIdx.x;
  const int lane = tid & 31;
  const int wid  = tid >> 5;
  const int gw   = blockIdx.x * 8 + wid;   // 0..63
  const int m0   = (gw & 1) * 16;
  const int n0   = (gw >> 1) * 64;
  const int half = lane >> 4;
  const int nlo  = lane & 15;
  const int row  = m0 + nlo;

  v8f acc[4] = {};
  for (int k0 = 0; k0 < Ktot; k0 += 32) {
    const _Float16* ap = (k0 < KX) ? (aX + (size_t)row * strideX + k0)
                                   : (aH + (size_t)row * KH + (k0 - KX));
    const v16h a = load_a_frag(ap, half);
#pragma unroll
    for (int nt = 0; nt < 4; ++nt) {
      const v16h b = load_b_frag(Wt + (size_t)(n0 + nt * 16 + nlo) * Ktot + k0 + half * 16);
      acc[nt] = __builtin_amdgcn_wmma_f32_16x16x32_f16(false, a, false, b, (short)0,
                                                       acc[nt], false, false);
    }
  }
#pragma unroll
  for (int nt = 0; nt < 4; ++nt) {
    const int n  = n0 + nt * 16 + nlo;
    const float bb = bias[n];
#pragma unroll
    for (int r = 0; r < 8; ++r) {
      const int m = m0 + r + 8 * half;
      gates[(size_t)m * 2048 + n] = acc[nt][r] + bb;
    }
  }
}

// ---------------------------------------------------------------------------
// Kernel 4: LSTM elementwise (gates order: [f | i | g | o], cols of 512)
// ---------------------------------------------------------------------------
__global__ void lstm_ew_kernel(const float* __restrict__ gates,
                               float* __restrict__ c, _Float16* __restrict__ h) {
  const int i = blockIdx.x * 256 + threadIdx.x;
  if (i >= 32 * 512) return;
  const int b = i >> 9, j = i & 511;
  const float* g = gates + (size_t)b * 2048;
  const float fg = g[j], ig = g[512 + j], gg = g[1024 + j], og = g[1536 + j];
  const float sf = 1.f / (1.f + __expf(-fg));
  const float si = 1.f / (1.f + __expf(-ig));
  const float so = 1.f / (1.f + __expf(-og));
  const float c2 = sf * c[i] + si * tanhf(gg);
  const float h2 = so * tanhf(c2);
  c[i] = c2;
  h[i] = (_Float16)h2;
}

// ---------------------------------------------------------------------------
// Helpers: weight conversion / init / classifier
// ---------------------------------------------------------------------------
__global__ void convert_transpose_kernel(const float* __restrict__ in,
                                         _Float16* __restrict__ out, int K, int N) {
  const int i = blockIdx.x * 256 + threadIdx.x;
  if (i >= K * N) return;
  const int k = i / N, n = i % N;
  out[(size_t)n * K + k] = (_Float16)in[i];
}

__global__ void convert_kernel(const float* __restrict__ in, _Float16* __restrict__ out, int n) {
  const int i = blockIdx.x * 256 + threadIdx.x;
  if (i < n) out[i] = (_Float16)in[i];
}

__global__ void init_state_kernel(_Float16* h0, _Float16* h1, float* c0, float* c1) {
  const int i = blockIdx.x * 256 + threadIdx.x;
  if (i < 32 * 512) {
    h0[i] = (_Float16)0.f;
    h1[i] = (_Float16)0.f;
    c0[i] = 0.f;
    c1[i] = 0.f;
  }
}

__global__ void classifier_kernel(const _Float16* __restrict__ h,
                                  const float* __restrict__ Wc,
                                  const float* __restrict__ bc,
                                  float* __restrict__ out) {
  const int t = threadIdx.x;
  if (t >= 32 * 6) return;
  const int b = t / 6, k = t % 6;
  float acc = bc[k];
  for (int j = 0; j < 512; ++j)
    acc = fmaf((float)h[(size_t)b * 512 + j], Wc[j * 6 + k], acc);
  out[b * 6 + k] = acc;
}

// ---------------------------------------------------------------------------
extern "C" void kernel_launch(void* const* d_in, const int* in_sizes, int n_in,
                              void* d_out, int out_size, void* d_ws, size_t ws_size,
                              hipStream_t stream) {
  (void)in_sizes; (void)n_in; (void)out_size; (void)ws_size;

  const float* x   = (const float*)d_in[0];   // (32,32,1,64,64)
  const float* w1  = (const float*)d_in[1];   // (16,1,3,3)
  const float* b1  = (const float*)d_in[2];
  const float* w2  = (const float*)d_in[3];   // (32,16,3,3)
  const float* b2  = (const float*)d_in[4];
  const float* w3  = (const float*)d_in[5];   // (256,32,3,3) -> (256,288) N-major
  const float* b3  = (const float*)d_in[6];
  const float* W0  = (const float*)d_in[7];   // (768,2048)
  const float* bg0 = (const float*)d_in[8];
  const float* W1  = (const float*)d_in[9];   // (1024,2048)
  const float* bg1 = (const float*)d_in[10];
  const float* Wc  = (const float*)d_in[11];  // (512,6)
  const float* bc  = (const float*)d_in[12];
  float* out = (float*)d_out;                 // (32,6)

  char* ws = (char*)d_ws;
  size_t off = 0;
  auto alloc = [&](size_t bytes) -> void* {
    void* p = ws + off;
    off = (off + bytes + 255) & ~(size_t)255;
    return p;
  };
  _Float16* Sbuf = (_Float16*)alloc((size_t)1024 * 288 * 2);
  _Float16* embh = (_Float16*)alloc((size_t)1024 * 256 * 2);
  _Float16* W3h  = (_Float16*)alloc((size_t)256 * 288 * 2);
  _Float16* W0t  = (_Float16*)alloc((size_t)2048 * 768 * 2);
  _Float16* W1t  = (_Float16*)alloc((size_t)2048 * 1024 * 2);
  float*    gts  = (float*)alloc((size_t)32 * 2048 * 4);
  _Float16* h0h  = (_Float16*)alloc((size_t)32 * 512 * 2);
  _Float16* h1h  = (_Float16*)alloc((size_t)32 * 512 * 2);
  float*    c0   = (float*)alloc((size_t)32 * 512 * 4);
  float*    c1   = (float*)alloc((size_t)32 * 512 * 4);

  // state + weight preparation
  init_state_kernel<<<64, 256, 0, stream>>>(h0h, h1h, c0, c1);
  convert_transpose_kernel<<<(768 * 2048 + 255) / 256, 256, 0, stream>>>(W0, W0t, 768, 2048);
  convert_transpose_kernel<<<(1024 * 2048 + 255) / 256, 256, 0, stream>>>(W1, W1t, 1024, 2048);
  convert_kernel<<<(256 * 288 + 255) / 256, 256, 0, stream>>>(w3, W3h, 256 * 288);

  // conv pipeline fused per frame (z1/z2 live only in LDS): 112 KB dynamic LDS
  frames_kernel<<<1024, 256, (4096 + 16384 + 8192) * sizeof(float), stream>>>(
      x, w1, b1, w2, b2, Sbuf);

  // emb = (S/256) @ W3^T + b3   (WMMA)
  emb_gemm_kernel<<<32, 256, 0, stream>>>(Sbuf, W3h, b3, embh);

  // 2-layer LSTM over T=32 timesteps (WMMA gate GEMMs)
  for (int t = 0; t < 32; ++t) {
    // layer 0: A = [emb_t (32x256) | h0 (32x512)], K = 768
    gate_gemm_kernel<<<8, 256, 0, stream>>>(embh + (size_t)t * 256, 32 * 256, 256,
                                            h0h, 512, W0t, 768, bg0, gts);
    lstm_ew_kernel<<<64, 256, 0, stream>>>(gts, c0, h0h);
    // layer 1: A = [h0 (32x512) | h1 (32x512)], K = 1024
    gate_gemm_kernel<<<8, 256, 0, stream>>>(h0h, 512, 512,
                                            h1h, 512, W1t, 1024, bg1, gts);
    lstm_ew_kernel<<<64, 256, 0, stream>>>(gts, c1, h1h);
  }

  // logits = h1 @ Wc + bc
  classifier_kernel<<<1, 192, 0, stream>>>(h1h, Wc, bc, out);
}
